// Tacotron2_83597243449407
// MI455X (gfx1250) — matrix operand order repaired and checked by reading, not compiled, not run on hardware
//
#include <hip/hip_runtime.h>
#include <hip/hip_bf16.h>
#include <math.h>

typedef __attribute__((ext_vector_type(16))) _Float16 v16h;
typedef __attribute__((ext_vector_type(8)))  float    v8f;

// ---------------------------------------------------------------------------
// Generic strided/batched GEMM: C[m,n] (+)= sum_k A[m,k]*B[k,n], f16 WMMA,
// f32 accumulate. One wave per 16x16 C tile. Out-of-range M rows / N cols are
// loaded from clamped addresses and masked only at the store; only the K tail
// needs zero-fill. act: 0=none, 1=relu, 2=tanh. bias added at store.
// ---------------------------------------------------------------------------
__device__ __forceinline__ void gemm_store(v8f acc, float* __restrict__ C,
                                           long long rsC, long long csC,
                                           const float* __restrict__ bias,
                                           int m0, int n0, int lm, int half,
                                           int M, int N, int act) {
#pragma unroll
    for (int r = 0; r < 8; ++r) {
        int row = m0 + r + half * 8;
        int col = n0 + lm;
        if (row < M && col < N) {
            float v = acc[r];
            if (bias) v += bias[col];
            if (act == 1) v = fmaxf(v, 0.0f);
            else if (act == 2) v = tanhf(v);
            C[(long long)row * rsC + (long long)col * csC] = v;
        }
    }
}

__global__ __launch_bounds__(32)
void gemm_wmma_kernel(const float* __restrict__ A, long long rsA, long long csA, long long sAb,
                      const float* __restrict__ Bm, long long rsB, long long csB, long long sBb,
                      float* __restrict__ C, long long rsC, long long csC, long long sCb,
                      const float* __restrict__ bias,
                      int M, int N, int K, int accum, int act)
{
    const int lane = threadIdx.x;
    A  += (long long)blockIdx.z * sAb;
    Bm += (long long)blockIdx.z * sBb;
    C  += (long long)blockIdx.z * sCb;
    const int m0 = blockIdx.y * 16, n0 = blockIdx.x * 16;
    const int lm = lane & 15, half = lane >> 4;

    const int am = m0 + lm;
    const int bn = n0 + lm;
    const int amc = am < M ? am : (M - 1);
    const int bnc = bn < N ? bn : (N - 1);
    const float* Arow = A  + (long long)amc * rsA;
    const float* Bcol = Bm + (long long)bnc * csB;

    v8f acc;
#pragma unroll
    for (int r = 0; r < 8; ++r) {
        int row = m0 + r + half * 8;
        int col = n0 + lm;
        acc[r] = (accum && row < M && col < N) ? C[(long long)row * rsC + (long long)col * csC] : 0.0f;
    }

    const int Kfull = K & ~31;
    int kk = 0;
#pragma unroll 2
    for (; kk < Kfull; kk += 32) {
        const float* Ak = Arow + (long long)kk * csA;
        const float* Bk = Bcol + (long long)kk * rsB;
        v16h af, bf;
#pragma unroll
        for (int i = 0; i < 16; ++i) {
            int j = i >> 1, p = i & 1;
            int kloc = (j < 4 ? (j * 2 + p) : (16 + (j - 4) * 2 + p)) + half * 8;  // 16-bit A 16x32 layout
            af[i] = (_Float16)Ak[(long long)kloc * csA];
        }
#pragma unroll
        for (int i = 0; i < 16; ++i) {
            int kloc = i + (i >= 8 ? 8 : 0) + half * 8;                            // B 32x16 layout (same k order as A)
            bf[i] = (_Float16)Bk[(long long)kloc * rsB];
        }
        acc = __builtin_amdgcn_wmma_f32_16x16x32_f16(false, af, false, bf, (short)0, acc, false, false);
    }
    if (kk < K) {  // K tail: only k-overflow must be zeroed (clamped, branch-free loads)
        v16h af, bf;
#pragma unroll
        for (int i = 0; i < 16; ++i) {
            int j = i >> 1, p = i & 1;
            int kloc = (j < 4 ? (j * 2 + p) : (16 + (j - 4) * 2 + p)) + half * 8;
            int k = kk + kloc;
            int kc = k < K ? k : (K - 1);
            float av = Arow[(long long)kc * csA];
            af[i] = (_Float16)(k < K ? av : 0.0f);
        }
#pragma unroll
        for (int i = 0; i < 16; ++i) {
            int k = kk + i + (i >= 8 ? 8 : 0) + half * 8;
            int kc = k < K ? k : (K - 1);
            float bv = Bcol[(long long)kc * rsB];
            bf[i] = (_Float16)(k < K ? bv : 0.0f);
        }
        acc = __builtin_amdgcn_wmma_f32_16x16x32_f16(false, af, false, bf, (short)0, acc, false, false);
    }
    gemm_store(acc, C, rsC, csC, bias, m0, n0, lm, half, M, N, act);
}

// Specialization for csA==1 && rsB==1 (all LSTM/attention/prenet GEMMs):
// fragment elements are contiguous per lane -> float2 (8B-aligned) loads,
// halving the load count per WMMA.
__global__ __launch_bounds__(32)
void gemm_wmma_vec_kernel(const float* __restrict__ A, long long rsA, long long sAb,
                          const float* __restrict__ Bm, long long csB, long long sBb,
                          float* __restrict__ C, long long rsC, long long csC, long long sCb,
                          const float* __restrict__ bias,
                          int M, int N, int K, int accum, int act)
{
    const int lane = threadIdx.x;
    A  += (long long)blockIdx.z * sAb;
    Bm += (long long)blockIdx.z * sBb;
    C  += (long long)blockIdx.z * sCb;
    const int m0 = blockIdx.y * 16, n0 = blockIdx.x * 16;
    const int lm = lane & 15, half = lane >> 4;

    const int am = m0 + lm;
    const int bn = n0 + lm;
    const int amc = am < M ? am : (M - 1);
    const int bnc = bn < N ? bn : (N - 1);
    const float* Arow = A  + (long long)amc * rsA;
    const float* Bcol = Bm + (long long)bnc * csB;

    v8f acc;
#pragma unroll
    for (int r = 0; r < 8; ++r) {
        int row = m0 + r + half * 8;
        int col = n0 + lm;
        acc[r] = (accum && row < M && col < N) ? C[(long long)row * rsC + (long long)col * csC] : 0.0f;
    }

    const int Kfull = K & ~31;
    int kk = 0;
#pragma unroll 2
    for (; kk < Kfull; kk += 32) {
        const float2* Ak0 = (const float2*)(Arow + kk + half * 8);        // af[0..7]
        const float2* Ak1 = (const float2*)(Arow + kk + 16 + half * 8);   // af[8..15]
        const float2* Bk0 = (const float2*)(Bcol + kk + half * 8);        // bf[0..7]
        const float2* Bk1 = (const float2*)(Bcol + kk + 16 + half * 8);   // bf[8..15]
        v16h af, bf;
#pragma unroll
        for (int q = 0; q < 4; ++q) {
            float2 a0 = Ak0[q];
            af[q * 2 + 0] = (_Float16)a0.x;
            af[q * 2 + 1] = (_Float16)a0.y;
            float2 a1 = Ak1[q];
            af[8 + q * 2 + 0] = (_Float16)a1.x;
            af[8 + q * 2 + 1] = (_Float16)a1.y;
        }
#pragma unroll
        for (int q = 0; q < 4; ++q) {
            float2 b0 = Bk0[q];
            bf[q * 2 + 0] = (_Float16)b0.x;
            bf[q * 2 + 1] = (_Float16)b0.y;
            float2 b1 = Bk1[q];
            bf[8 + q * 2 + 0] = (_Float16)b1.x;
            bf[8 + q * 2 + 1] = (_Float16)b1.y;
        }
        acc = __builtin_amdgcn_wmma_f32_16x16x32_f16(false, af, false, bf, (short)0, acc, false, false);
    }
    if (kk < K) {
        v16h af, bf;
#pragma unroll
        for (int i = 0; i < 16; ++i) {
            int j = i >> 1, p = i & 1;
            int kloc = (j < 4 ? (j * 2 + p) : (16 + (j - 4) * 2 + p)) + half * 8;
            int k = kk + kloc;
            int kc = k < K ? k : (K - 1);
            float av = Arow[kc];
            af[i] = (_Float16)(k < K ? av : 0.0f);
        }
#pragma unroll
        for (int i = 0; i < 16; ++i) {
            int k = kk + i + (i >= 8 ? 8 : 0) + half * 8;
            int kc = k < K ? k : (K - 1);
            float bv = Bcol[kc];
            bf[i] = (_Float16)(k < K ? bv : 0.0f);
        }
        acc = __builtin_amdgcn_wmma_f32_16x16x32_f16(false, af, false, bf, (short)0, acc, false, false);
    }
    gemm_store(acc, C, rsC, csC, bias, m0, n0, lm, half, M, N, act);
}

// ---------------------------------------------------------------------------
// Elementwise / small kernels
// ---------------------------------------------------------------------------
__global__ void fill_kernel(float* p, float v, long long n) {
    long long i = (long long)blockIdx.x * blockDim.x + threadIdx.x;
    if (i < n) p[i] = v;
}

__global__ void embed_kernel(const int* xs, const float* ew, float* out, int B, int T, int E) {
    long long i = (long long)blockIdx.x * blockDim.x + threadIdx.x;
    long long tot = (long long)B * E * T;
    if (i >= tot) return;
    int t = (int)(i % T);
    int e = (int)((i / T) % E);
    int b = (int)(i / ((long long)T * E));
    out[i] = ew[(long long)xs[b * T + t] * E + e];
}

__global__ void pad1d_kernel(const float* x, float* xp, int B, int C, int T, int P) {
    long long i = (long long)blockIdx.x * blockDim.x + threadIdx.x;
    int W = T + 2 * P;
    long long tot = (long long)B * C * W;
    if (i >= tot) return;
    int tp = (int)(i % W);
    int c  = (int)((i / W) % C);
    int b  = (int)(i / ((long long)W * C));
    int t = tp - P;
    xp[i] = (t >= 0 && t < T) ? x[((long long)b * C + c) * T + t] : 0.0f;
}

__global__ void bn_act_kernel(const float* x, float* y, const float* g, const float* bb,
                              int B, int C, int T, int act) {
    long long i = (long long)blockIdx.x * blockDim.x + threadIdx.x;
    long long tot = (long long)B * C * T;
    if (i >= tot) return;
    int c = (int)((i / T) % C);
    float s = g[c] * rsqrtf(1.0f + 1e-5f);
    float v = x[i] * s + bb[c];
    if (act == 1) v = fmaxf(v, 0.0f);
    else if (act == 2) v = tanhf(v);
    y[i] = v;
}

// (B,E,T) -> (T*B, E) rows (t*B+b)
__global__ void to_tb_kernel(const float* x, float* y, int B, int E, int T) {
    long long i = (long long)blockIdx.x * blockDim.x + threadIdx.x;
    long long tot = (long long)T * B * E;
    if (i >= tot) return;
    int e = (int)(i % E);
    int b = (int)((i / E) % B);
    int t = (int)(i / ((long long)E * B));
    y[i] = x[((long long)b * E + e) * T + t];
}

__global__ void lstm_cell_kernel(const float* gA, const float* gB, const float* bias,
                                 float* h, float* c, float* hstore, int B, int H) {
    long long i = (long long)blockIdx.x * blockDim.x + threadIdx.x;
    if (i >= (long long)B * H) return;
    int j = (int)(i % H);
    int b = (int)(i / H);
    long long base = (long long)b * 4 * H;
    float gi = gA[base + j], gf = gA[base + H + j], gg = gA[base + 2 * H + j], go = gA[base + 3 * H + j];
    if (gB) {
        gi += gB[base + j]; gf += gB[base + H + j]; gg += gB[base + 2 * H + j]; go += gB[base + 3 * H + j];
    }
    gi += bias[j]; gf += bias[H + j]; gg += bias[2 * H + j]; go += bias[3 * H + j];
    float si = 1.0f / (1.0f + expf(-gi));
    float sf = 1.0f / (1.0f + expf(-gf));
    float so = 1.0f / (1.0f + expf(-go));
    float cn = sf * c[i] + si * tanhf(gg);
    float hn = so * tanhf(cn);
    c[i] = cn; h[i] = hn;
    if (hstore) hstore[i] = hn;
}

__global__ void concat_hs_kernel(const float* hf, const float* hb, float* hs, int B, int T, int H) {
    long long i = (long long)blockIdx.x * blockDim.x + threadIdx.x;
    long long tot = (long long)B * T * 2 * H;
    if (i >= tot) return;
    int ch = (int)(i % (2 * H));
    int t  = (int)((i / (2 * H)) % T);
    int b  = (int)(i / ((long long)2 * H * T));
    hs[i] = (ch < H) ? hf[((long long)t * B + b) * H + ch] : hb[((long long)t * B + b) * H + ch - H];
}

__global__ void prev_kernel(const float* ys, float* Pm, int B, int L, int od) {
    long long i = (long long)blockIdx.x * blockDim.x + threadIdx.x;
    long long tot = (long long)L * B * od;
    if (i >= tot) return;
    int d = (int)(i % od);
    int b = (int)((i / od) % B);
    int l = (int)(i / ((long long)od * B));
    Pm[i] = (l == 0) ? 0.0f : ys[((long long)b * L + (l - 1)) * od + d];
}

// location conv, output transposed: acT[(b*T+t)*F + f]
__global__ void locconv_kernel(const float* w, const float* lcw, float* acT, int B, int T, int F, int Kc) {
    long long i = (long long)blockIdx.x * blockDim.x + threadIdx.x;
    long long tot = (long long)B * T * F;
    if (i >= tot) return;
    int f = (int)(i % F);
    int t = (int)((i / F) % T);
    int b = (int)(i / ((long long)F * T));
    int P = (Kc - 1) / 2;
    float s = 0.0f;
    for (int k = 0; k < Kc; ++k) {
        int tt = t + k - P;
        if (tt >= 0 && tt < T) s += lcw[f * Kc + k] * w[b * T + tt];
    }
    acT[i] = s;
}

__global__ __launch_bounds__(32)
void edot_kernel(const float* acproj, const float* encp, const float* dz,
                 const float* gw, const float* gb, float* e, int T, int A) {
    int row = blockIdx.x;
    int b = row / T;
    int lane = threadIdx.x;
    float s = 0.0f;
    for (int i = lane; i < A; i += 32) {
        float v = tanhf(acproj[(long long)row * A + i] + encp[(long long)row * A + i] + dz[(long long)b * A + i]);
        s += v * gw[i];
    }
    for (int off = 16; off; off >>= 1) s += __shfl_down(s, off);
    if (lane == 0) e[row] = s + gb[0];
}

__global__ void softmax_kernel(const float* e, float* w, int T) {
    __shared__ float red[128];
    int b = blockIdx.x, tid = threadIdx.x;
    float m = -1e30f;
    for (int t = tid; t < T; t += blockDim.x) m = fmaxf(m, 2.0f * e[b * T + t]);
    red[tid] = m; __syncthreads();
    for (int s = 64; s; s >>= 1) { if (tid < s) red[tid] = fmaxf(red[tid], red[tid + s]); __syncthreads(); }
    m = red[0]; __syncthreads();
    float sum = 0.0f;
    for (int t = tid; t < T; t += blockDim.x) sum += expf(2.0f * e[b * T + t] - m);
    red[tid] = sum; __syncthreads();
    for (int s = 64; s; s >>= 1) { if (tid < s) red[tid] += red[tid + s]; __syncthreads(); }
    sum = red[0];
    for (int t = tid; t < T; t += blockDim.x) w[b * T + t] = expf(2.0f * e[b * T + t] - m) / sum;
}

__global__ void ctx_kernel(const float* w, const float* hs, float* ctx, int B, int T, int D) {
    long long i = (long long)blockIdx.x * blockDim.x + threadIdx.x;
    if (i >= (long long)B * D) return;
    int d = (int)(i % D);
    int b = (int)(i / D);
    float s = 0.0f;
    for (int t = 0; t < T; ++t) s += w[b * T + t] * hs[((long long)b * T + t) * D + d];
    ctx[i] = s;
}

__global__ __launch_bounds__(32)
void prob_kernel(const float* z1, const float* pw, const float* pb, float* outp, int Dz, int L) {
    int b = blockIdx.x, lane = threadIdx.x;
    float s = 0.0f;
    for (int i = lane; i < Dz; i += 32) s += z1[(long long)b * Dz + i] * pw[i];
    for (int off = 16; off; off >>= 1) s += __shfl_down(s, off);
    if (lane == 0) outp[(long long)b * L] = s + pb[0];
}

__global__ void final_kernel(const float* outs_bdl, const float* pfin, float* out0, float* out1,
                             int B, int L, int od) {
    long long i = (long long)blockIdx.x * blockDim.x + threadIdx.x;
    long long tot = (long long)B * L * od;
    if (i >= tot) return;
    int d = (int)(i % od);
    int l = (int)((i / od) % L);
    int b = (int)(i / ((long long)od * L));
    float o = outs_bdl[((long long)b * od + d) * L + l];
    out1[i] = o;
    out0[i] = o + pfin[((long long)b * od + d) * L + l];
}

// ---------------------------------------------------------------------------
// Host side
// ---------------------------------------------------------------------------
static inline void gemm(hipStream_t st,
                        const float* A, long long rsA, long long csA, long long sAb,
                        const float* Bm, long long rsB, long long csB, long long sBb,
                        float* C, long long rsC, long long csC, long long sCb,
                        const float* bias, int M, int N, int K, int nb, int accum, int act) {
    dim3 grid((N + 15) / 16, (M + 15) / 16, nb);
    if (csA == 1 && rsB == 1) {
        gemm_wmma_vec_kernel<<<grid, dim3(32), 0, st>>>(A, rsA, sAb, Bm, csB, sBb,
                                                        C, rsC, csC, sCb, bias, M, N, K, accum, act);
    } else {
        gemm_wmma_kernel<<<grid, dim3(32), 0, st>>>(A, rsA, csA, sAb, Bm, rsB, csB, sBb,
                                                    C, rsC, csC, sCb, bias, M, N, K, accum, act);
    }
}

static inline unsigned eg(long long n) { return (unsigned)((n + 255) / 256); }

extern "C" void kernel_launch(void* const* d_in, const int* in_sizes, int n_in,
                              void* d_out, int out_size, void* d_ws, size_t ws_size,
                              hipStream_t stream) {
    const int B = 16, T = 100, L = 200, E = 512, D = 1024, A = 512, OD = 80, H = 256;
    const int F = 10, KC = 201;

    const int*   xs        = (const int*)d_in[0];
    const float* embed_w   = (const float*)d_in[3];
    const float* enc_conv_w= (const float*)d_in[4];
    const float* enc_bn_g  = (const float*)d_in[5];
    const float* enc_bn_b  = (const float*)d_in[6];
    const float* blstm_wih = (const float*)d_in[7];
    const float* blstm_whh = (const float*)d_in[8];
    const float* blstm_b   = (const float*)d_in[9];
    const float* att_enc_w = (const float*)d_in[10];
    const float* att_enc_b = (const float*)d_in[11];
    const float* att_dec_w = (const float*)d_in[12];
    const float* att_att_w = (const float*)d_in[13];
    const float* loc_conv_w= (const float*)d_in[14];
    const float* gvec_w    = (const float*)d_in[15];
    const float* gvec_b    = (const float*)d_in[16];
    const float* prenet_w1 = (const float*)d_in[17];
    const float* prenet_w2 = (const float*)d_in[18];
    const float* lstm0_wih = (const float*)d_in[19];
    const float* lstm0_whh = (const float*)d_in[20];
    const float* lstm0_b   = (const float*)d_in[21];
    const float* lstm1_wih = (const float*)d_in[22];
    const float* lstm1_whh = (const float*)d_in[23];
    const float* lstm1_b   = (const float*)d_in[24];
    const float* feat_out_w= (const float*)d_in[25];
    const float* prob_out_w= (const float*)d_in[26];
    const float* prob_out_b= (const float*)d_in[27];
    const float* post_w_in = (const float*)d_in[28];
    const float* post_g_in = (const float*)d_in[29];
    const float* post_b_in = (const float*)d_in[30];
    const float* post_w_mid= (const float*)d_in[31];
    const float* post_g_mid= (const float*)d_in[32];
    const float* post_b_mid= (const float*)d_in[33];
    const float* post_w_out= (const float*)d_in[34];
    const float* post_g_out= (const float*)d_in[35];
    const float* post_b_out= (const float*)d_in[36];
    const float* ys        = (const float*)d_in[37];

    float* out_post  = (float*)d_out;                       // (B,L,OD)
    float* out_feats = out_post + (size_t)B * L * OD;       // (B,L,OD)
    float* out_probs = out_feats + (size_t)B * L * OD;      // (B,L)

    // ---- workspace carve ----
    float* ws = (float*)d_ws;
    size_t off = 0;
    auto carve = [&](size_t n) { float* p = ws + off; off += n; return p; };
    float* xA      = carve((size_t)B * E * T);
    float* xB      = carve((size_t)B * E * T);
    float* padbuf  = carve((size_t)B * 512 * (L + 4));     // shared enc/post pad
    float* convout = carve((size_t)B * 512 * L);           // enc conv out / post ping2
    float* X_tb    = carve((size_t)T * B * E);
    float* gx      = carve((size_t)T * B * 4 * H);
    float* gtmp    = carve((size_t)B * 4 * H);
    float* hstate  = carve((size_t)B * H);
    float* cstate  = carve((size_t)B * H);
    float* hf      = carve((size_t)T * B * H);
    float* hb      = carve((size_t)T * B * H);
    float* hs      = carve((size_t)B * T * 2 * H);
    float* encp    = carve((size_t)B * T * A);
    float* Pm      = carve((size_t)L * B * OD);
    float* P1      = carve((size_t)L * B * 256);
    float* P2      = carve((size_t)L * B * 256);
    float* att_w   = carve((size_t)B * T);
    float* acT     = carve((size_t)B * T * F);
    float* acproj  = carve((size_t)B * T * A);
    float* dz      = carve((size_t)B * A);
    float* evec    = carve((size_t)B * T);
    float* ctxv    = carve((size_t)B * (2 * H));
    float* gates0  = carve((size_t)B * 4 * D);
    float* gates1  = carve((size_t)B * 4 * D);
    float* z0      = carve((size_t)B * D);
    float* c0      = carve((size_t)B * D);
    float* z1      = carve((size_t)B * D);
    float* c1      = carve((size_t)B * D);
    float* outs_bdl= carve((size_t)B * OD * L);
    float* pp1     = carve((size_t)B * 512 * L);
    float* pfin    = carve((size_t)B * OD * L);
    (void)ws_size; (void)n_in; (void)in_sizes; (void)out_size;

    // ================= Encoder =================
    embed_kernel<<<eg((long long)B * E * T), 256, 0, stream>>>(xs, embed_w, xA, B, T, E);
    float* src = xA; float* dst = xB;
    for (int layer = 0; layer < 3; ++layer) {
        pad1d_kernel<<<eg((long long)B * E * (T + 4)), 256, 0, stream>>>(src, padbuf, B, E, T, 2);
        for (int k = 0; k < 5; ++k) {
            // Y[b](Cout,T) += W_k(Cout,Cin) @ Xpad[b](Cin,T) shifted by k
            gemm(stream,
                 enc_conv_w + (size_t)layer * E * E * 5 + k, (long long)E * 5, 5, 0,
                 padbuf + k, T + 4, 1, (long long)E * (T + 4),
                 convout, T, 1, (long long)E * T,
                 nullptr, E, T, E, B, k > 0, 0);
        }
        bn_act_kernel<<<eg((long long)B * E * T), 256, 0, stream>>>(
            convout, dst, enc_bn_g + layer * E, enc_bn_b + layer * E, B, E, T, 1);
        float* tmp = src; src = dst; dst = tmp;
    }
    // (B,E,T) -> (T*B, E)
    to_tb_kernel<<<eg((long long)T * B * E), 256, 0, stream>>>(src, X_tb, B, E, T);

    // ================= BLSTM =================
    for (int dir = 0; dir < 2; ++dir) {
        const float* wih = blstm_wih + (size_t)dir * 4 * H * E;
        const float* whh = blstm_whh + (size_t)dir * 4 * H * H;
        const float* bb  = blstm_b + (size_t)dir * 4 * H;
        // gx = X_tb @ wih^T  (T*B, 4H)
        gemm(stream, X_tb, E, 1, 0, wih, 1, E, 0, gx, 4 * H, 1, 0,
             nullptr, T * B, 4 * H, E, 1, 0, 0);
        fill_kernel<<<eg((long long)B * H), 256, 0, stream>>>(hstate, 0.0f, (long long)B * H);
        fill_kernel<<<eg((long long)B * H), 256, 0, stream>>>(cstate, 0.0f, (long long)B * H);
        float* hdir = (dir == 0) ? hf : hb;
        for (int s = 0; s < T; ++s) {
            int t = (dir == 0) ? s : (T - 1 - s);
            gemm(stream, hstate, H, 1, 0, whh, 1, H, 0, gtmp, 4 * H, 1, 0,
                 nullptr, B, 4 * H, H, 1, 0, 0);
            lstm_cell_kernel<<<eg((long long)B * H), 256, 0, stream>>>(
                gtmp, gx + (size_t)t * B * 4 * H, bb, hstate, cstate,
                hdir + (size_t)t * B * H, B, H);
        }
    }
    concat_hs_kernel<<<eg((long long)B * T * 2 * H), 256, 0, stream>>>(hf, hb, hs, B, T, H);

    // enc_proj = hs @ att_enc_w^T + b   (B*T, A)
    gemm(stream, hs, 2 * H, 1, 0, att_enc_w, 1, 2 * H, 0, encp, A, 1, 0,
         att_enc_b, B * T, A, 2 * H, 1, 0, 0);

    // ================= Prenet precompute (teacher forcing: inputs known) ====
    prev_kernel<<<eg((long long)L * B * OD), 256, 0, stream>>>(ys, Pm, B, L, OD);
    gemm(stream, Pm, OD, 1, 0, prenet_w1, 1, OD, 0, P1, 256, 1, 0,
         nullptr, L * B, 256, OD, 1, 0, 1);
    gemm(stream, P1, 256, 1, 0, prenet_w2, 1, 256, 0, P2, 256, 1, 0,
         nullptr, L * B, 256, 256, 1, 0, 1);

    // ================= Decoder loop =================
    fill_kernel<<<eg((long long)B * T), 256, 0, stream>>>(att_w, 1.0f / T, (long long)B * T);
    fill_kernel<<<eg((long long)B * D), 256, 0, stream>>>(z0, 0.0f, (long long)B * D);
    fill_kernel<<<eg((long long)B * D), 256, 0, stream>>>(c0, 0.0f, (long long)B * D);
    fill_kernel<<<eg((long long)B * D), 256, 0, stream>>>(z1, 0.0f, (long long)B * D);
    fill_kernel<<<eg((long long)B * D), 256, 0, stream>>>(c1, 0.0f, (long long)B * D);

    for (int t = 0; t < L; ++t) {
        // location-aware attention
        locconv_kernel<<<eg((long long)B * T * F), 256, 0, stream>>>(att_w, loc_conv_w, acT, B, T, F, KC);
        gemm(stream, acT, F, 1, 0, att_att_w, 1, F, 0, acproj, A, 1, 0,
             nullptr, B * T, A, F, 1, 0, 0);
        gemm(stream, z0, D, 1, 0, att_dec_w, 1, D, 0, dz, A, 1, 0,
             nullptr, B, A, D, 1, 0, 0);
        edot_kernel<<<B * T, 32, 0, stream>>>(acproj, encp, dz, gvec_w, gvec_b, evec, T, A);
        softmax_kernel<<<B, 128, 0, stream>>>(evec, att_w, T);
        ctx_kernel<<<eg((long long)B * 2 * H), 256, 0, stream>>>(att_w, hs, ctxv, B, T, 2 * H);

        // LSTM0: gates = ctx@Wih[:, :512]^T + p@Wih[:, 512:]^T + z0@Whh^T
        gemm(stream, ctxv, 2 * H, 1, 0, lstm0_wih, 1, 2 * H + 256, 0,
             gates0, 4 * D, 1, 0, nullptr, B, 4 * D, 2 * H, 1, 0, 0);
        gemm(stream, P2 + (size_t)t * B * 256, 256, 1, 0, lstm0_wih + 2 * H, 1, 2 * H + 256, 0,
             gates0, 4 * D, 1, 0, nullptr, B, 4 * D, 256, 1, 1, 0);
        gemm(stream, z0, D, 1, 0, lstm0_whh, 1, D, 0,
             gates0, 4 * D, 1, 0, nullptr, B, 4 * D, D, 1, 1, 0);
        lstm_cell_kernel<<<eg((long long)B * D), 256, 0, stream>>>(
            gates0, nullptr, lstm0_b, z0, c0, nullptr, B, D);

        // LSTM1
        gemm(stream, z0, D, 1, 0, lstm1_wih, 1, D, 0,
             gates1, 4 * D, 1, 0, nullptr, B, 4 * D, D, 1, 0, 0);
        gemm(stream, z1, D, 1, 0, lstm1_whh, 1, D, 0,
             gates1, 4 * D, 1, 0, nullptr, B, 4 * D, D, 1, 1, 0);
        lstm_cell_kernel<<<eg((long long)B * D), 256, 0, stream>>>(
            gates1, nullptr, lstm1_b, z1, c1, nullptr, B, D);

        // feat out -> outs_bdl[:, :, t]   (strided C store)
        gemm(stream, z1, D, 1, 0, feat_out_w, 1, D, 0,
             outs_bdl + t, (long long)OD * L, L, 0,
             nullptr, B, OD, D, 1, 0, 0);
        prob_kernel<<<B, 32, 0, stream>>>(z1, prob_out_w, prob_out_b, out_probs + t, D, L);
    }

    // ================= Postnet =================
    // in: (B,80,L) -> (B,512,L), tanh
    pad1d_kernel<<<eg((long long)B * OD * (L + 4)), 256, 0, stream>>>(outs_bdl, padbuf, B, OD, L, 2);
    for (int k = 0; k < 5; ++k)
        gemm(stream, post_w_in + k, (long long)OD * 5, 5, 0,
             padbuf + k, L + 4, 1, (long long)OD * (L + 4),
             pp1, L, 1, (long long)512 * L,
             nullptr, 512, L, OD, B, k > 0, 0);
    bn_act_kernel<<<eg((long long)B * 512 * L), 256, 0, stream>>>(pp1, pp1, post_g_in, post_b_in, B, 512, L, 2);

    float* pa = pp1; float* pb = convout;
    for (int layer = 0; layer < 3; ++layer) {
        pad1d_kernel<<<eg((long long)B * 512 * (L + 4)), 256, 0, stream>>>(pa, padbuf, B, 512, L, 2);
        for (int k = 0; k < 5; ++k)
            gemm(stream, post_w_mid + (size_t)layer * 512 * 512 * 5 + k, (long long)512 * 5, 5, 0,
                 padbuf + k, L + 4, 1, (long long)512 * (L + 4),
                 pb, L, 1, (long long)512 * L,
                 nullptr, 512, L, 512, B, k > 0, 0);
        bn_act_kernel<<<eg((long long)B * 512 * L), 256, 0, stream>>>(
            pb, pb, post_g_mid + layer * 512, post_b_mid + layer * 512, B, 512, L, 2);
        float* tmp = pa; pa = pb; pb = tmp;
    }
    // out: (B,512,L) -> (B,80,L), linear BN
    pad1d_kernel<<<eg((long long)B * 512 * (L + 4)), 256, 0, stream>>>(pa, padbuf, B, 512, L, 2);
    for (int k = 0; k < 5; ++k)
        gemm(stream, post_w_out + k, (long long)512 * 5, 5, 0,
             padbuf + k, L + 4, 1, (long long)512 * (L + 4),
             pfin, L, 1, (long long)OD * L,
             nullptr, OD, L, 512, B, k > 0, 0);
    bn_act_kernel<<<eg((long long)B * OD * L), 256, 0, stream>>>(pfin, pfin, post_g_out, post_b_out, B, OD, L, 0);

    final_kernel<<<eg((long long)B * L * OD), 256, 0, stream>>>(outs_bdl, pfin, out_post, out_feats, B, L, OD);
}
